// PNANet_16793322128010
// MI455X (gfx1250) — matrix operand order repaired and checked by reading, not compiled, not run on hardware
//
#include <hip/hip_runtime.h>
#include <math.h>

#define N_NODES 32768
#define N_EDGES 524288
#define N_GRAPH 256
#define NODES_PER_GRAPH 128
#define DIM 64
#define LAYERS 4
#define EPS_F 1e-5f
#define AVG_D_LOG 2.8332133440562162f /* log(17.0) */

typedef __attribute__((ext_vector_type(16))) __bf16 v16bf;
typedef __attribute__((ext_vector_type(8)))  float  v8f;
typedef __attribute__((ext_vector_type(16))) unsigned short v16u16;
typedef __attribute__((ext_vector_type(8)))  unsigned short v8u16;

// float -> bf16 (round-to-nearest-even), stored as u16 to avoid relying on
// __bf16 arithmetic lowering.
__device__ __forceinline__ unsigned short f2u16(float f) {
  unsigned u = __builtin_bit_cast(unsigned, f);
  unsigned r = u + 0x7FFFu + ((u >> 16) & 1u);
  return (unsigned short)(r >> 16);
}

__device__ __forceinline__ void atom_add(float* p, float v) {
  __hip_atomic_fetch_add(p, v, __ATOMIC_RELAXED, __HIP_MEMORY_SCOPE_AGENT);
}
__device__ __forceinline__ void atom_max(float* p, float v) {
  __hip_atomic_fetch_max(p, v, __ATOMIC_RELAXED, __HIP_MEMORY_SCOPE_AGENT);
}
__device__ __forceinline__ void atom_min(float* p, float v) {
  __hip_atomic_fetch_min(p, v, __ATOMIC_RELAXED, __HIP_MEMORY_SCOPE_AGENT);
}

// ---------------------------------------------------------------------------
// Weight packing into WMMA B-fragment order.
// B (32x16, bf16): lane = n + 16*half; elems 0..7 -> K=half*8+i, 8..15 -> 16+half*8+(i-8).
// Fragment index (t, c): frag base = (t*NCHUNK + c)*512 ; lane*16 + elem.
// ---------------------------------------------------------------------------
__global__ void __launch_bounds__(256)
pack_wpre_kernel(const float* __restrict__ W, unsigned short* __restrict__ wpk) {
  int idx = blockIdx.x * 256 + threadIdx.x;          // L*4*6*512 = 49152
  if (idx >= LAYERS * 4 * 6 * 512) return;
  int elem = idx & 15;
  int lane = (idx >> 4) & 31;
  int q = idx >> 9;                                   // fragment index
  int c = q % 6; q /= 6;
  int t = q % 4; q /= 4;
  int l = q;
  int half = lane >> 4, n = lane & 15;
  int kg = c * 32 + (elem < 8 ? half * 8 + elem : 16 + half * 8 + (elem - 8)); // 0..191
  int ng = t * 16 + n;                                // 0..63
  wpk[idx] = f2u16(W[((size_t)l * 192 + kg) * 64 + ng]);
}

__global__ void __launch_bounds__(256)
pack_wpost_kernel(const float* __restrict__ W, unsigned short* __restrict__ wpk) {
  int idx = blockIdx.x * 256 + threadIdx.x;          // L*4*26*512 = 212992
  if (idx >= LAYERS * 4 * 26 * 512) return;
  int elem = idx & 15;
  int lane = (idx >> 4) & 31;
  int q = idx >> 9;
  int c = q % 26; q /= 26;
  int t = q % 4; q /= 4;
  int l = q;
  int half = lane >> 4, n = lane & 15;
  int kg = c * 32 + (elem < 8 ? half * 8 + elem : 16 + half * 8 + (elem - 8)); // 0..831
  int ng = t * 16 + n;
  wpk[idx] = f2u16(W[((size_t)l * 832 + kg) * 64 + ng]);
}

// ---------------------------------------------------------------------------
// Node embedding + bf16 mirror.
// ---------------------------------------------------------------------------
__global__ void __launch_bounds__(256)
embed_kernel(const int* __restrict__ h_tok, const float* __restrict__ emb_h,
             float* __restrict__ h, unsigned short* __restrict__ hb) {
  int idx = blockIdx.x * 256 + threadIdx.x;          // N*64
  int n = idx >> 6, f = idx & 63;
  float v = emb_h[h_tok[n] * 64 + f];
  h[idx] = v;
  hb[idx] = f2u16(v);
}

__global__ void __launch_bounds__(256)
deg_zero_kernel(float* __restrict__ deg) {
  int n = blockIdx.x * 256 + threadIdx.x;
  deg[n] = 0.f;
}

__global__ void __launch_bounds__(256)
deg_count_kernel(const int* __restrict__ dst, float* __restrict__ deg) {
  int e = blockIdx.x * 256 + threadIdx.x;
  atom_add(&deg[dst[e]], 1.0f);
}

__global__ void __launch_bounds__(256)
deg_scalars_kernel(const float* __restrict__ deg, float* __restrict__ amp,
                   float* __restrict__ att) {
  int n = blockIdx.x * 256 + threadIdx.x;
  float ld = logf(deg[n] + 1.0f);
  amp[n] = ld / AVG_D_LOG;
  att[n] = AVG_D_LOG / fmaxf(ld, 1e-6f);
}

// ---------------------------------------------------------------------------
// Per-layer accumulator init: s1=s2=0, mx=-inf-ish, mn=+inf-ish, bn[128]=0.
// ---------------------------------------------------------------------------
__global__ void __launch_bounds__(256)
layer_init_kernel(float* __restrict__ s1, float* __restrict__ s2,
                  float* __restrict__ mx, float* __restrict__ mn,
                  float* __restrict__ bn) {
  int idx = blockIdx.x * 256 + threadIdx.x;          // N*64
  s1[idx] = 0.f;
  s2[idx] = 0.f;
  mx[idx] = -3.0e38f;
  mn[idx] = 3.0e38f;
  if (idx < 128) bn[idx] = 0.f;
}

// ---------------------------------------------------------------------------
// Edge pretrans + scatter. One wave per 16-edge tile.
// A (16x32 bf16): lane = m + 16*half; elems 0..7 -> K=half*8+j, 8..15 -> 16+half*8+j.
// K layout: [h_src(0:64) | h_dst(64:128) | e(128:192)] -> 6 k-chunks.
// D (16x16 f32): VGPR r holds M = r + 8*half, N = lane&15.
// ---------------------------------------------------------------------------
__global__ void __launch_bounds__(256)
edge_kernel(const unsigned short* __restrict__ hb,       // [N,64] bf16
            const int* __restrict__ src, const int* __restrict__ dst,
            const int* __restrict__ e_tok, const float* __restrict__ emb_e,
            const unsigned short* __restrict__ wpk,      // layer's packed W_pre
            const float* __restrict__ b_pre,             // [64]
            float* __restrict__ s1, float* __restrict__ s2,
            float* __restrict__ mx, float* __restrict__ mn) {
  int wave = threadIdx.x >> 5;
  int lane = threadIdx.x & 31;
  int tile = blockIdx.x * 8 + wave;
  int ebase = tile * 16;
  int half = lane >> 4;
  int m = lane & 15;
  int edge = ebase + m;

  const unsigned short* rs = hb + (size_t)src[edge] * 64;
  const unsigned short* rd = hb + (size_t)dst[edge] * 64;
  const float* re = emb_e + e_tok[edge] * 64;

  v8f acc[4] = {};
#pragma unroll
  for (int c = 0; c < 6; ++c) {
    v16u16 a;
    if (c < 4) {
      const unsigned short* p = (c < 2 ? rs : rd) + (c & 1) * 32;
      v8u16 lo = *(const v8u16*)(p + half * 8);
      v8u16 hi = *(const v8u16*)(p + 16 + half * 8);
#pragma unroll
      for (int j = 0; j < 8; ++j) { a[j] = lo[j]; a[8 + j] = hi[j]; }
    } else {
      const float* p = re + (c - 4) * 32;
#pragma unroll
      for (int j = 0; j < 8; ++j) {
        a[j] = f2u16(p[half * 8 + j]);
        a[8 + j] = f2u16(p[16 + half * 8 + j]);
      }
    }
    v16bf av = __builtin_bit_cast(v16bf, a);
#pragma unroll
    for (int t = 0; t < 4; ++t) {
      v16u16 b = *(const v16u16*)(wpk + ((size_t)(t * 6 + c) * 32 + lane) * 16);
      v16bf bv = __builtin_bit_cast(v16bf, b);
      acc[t] = __builtin_amdgcn_wmma_f32_16x16x32_bf16(
          false, av, false, bv, (short)0, acc[t], false, false);
    }
  }

  // Scatter: msg -> sum / sum^2 / max / min keyed by dst.
#pragma unroll
  for (int r = 0; r < 8; ++r) {
    int e2 = ebase + r + 8 * half;
    size_t rowoff = (size_t)dst[e2] * 64;
#pragma unroll
    for (int t = 0; t < 4; ++t) {
      int feat = t * 16 + m;
      float v = acc[t][r] + b_pre[feat];
      atom_add(&s1[rowoff + feat], v);
      atom_add(&s2[rowoff + feat], v * v);
      atom_max(&mx[rowoff + feat], v);
      atom_min(&mn[rowoff + feat], v);
    }
  }
}

// ---------------------------------------------------------------------------
// Build base = [mean | max | min | std] per node (bf16 [N,256]).
// ---------------------------------------------------------------------------
__global__ void __launch_bounds__(256)
node_prep_kernel(const float* __restrict__ s1, const float* __restrict__ s2,
                 const float* __restrict__ mx, const float* __restrict__ mn,
                 const float* __restrict__ deg, unsigned short* __restrict__ base) {
  int idx = blockIdx.x * 256 + threadIdx.x;          // N*64
  int n = idx >> 6, f = idx & 63;
  float dg = deg[n];
  float ds = fmaxf(dg, 1.0f);
  float mean = s1[idx] / ds;
  float var = s2[idx] / ds - mean * mean;
  float sd = sqrtf(fmaxf(var, 0.0f) + EPS_F);
  float vmx = dg > 0.f ? mx[idx] : 0.f;
  float vmn = dg > 0.f ? mn[idx] : 0.f;
  size_t o = (size_t)n * 256 + f;
  base[o] = f2u16(mean);
  base[o + 64] = f2u16(vmx);
  base[o + 128] = f2u16(vmn);
  base[o + 192] = f2u16(sd);
}

// ---------------------------------------------------------------------------
// Node posttrans GEMM. One wave per 16-node tile.
// h_out = h@Wa + base@Wb + amp.*(base@Wc) + att.*(base@Wd) + b_post, then
// * snorm; accumulates per-feature BN sums.
// Packed W_post chunk map: c=0..1 -> Wa(h), 2..9 -> Wb, 10..17 -> Wc, 18..25 -> Wd.
// ---------------------------------------------------------------------------
__global__ void __launch_bounds__(256)
node_gemm_kernel(const unsigned short* __restrict__ hb,     // [N,64] bf16
                 const unsigned short* __restrict__ base,   // [N,256] bf16
                 const unsigned short* __restrict__ wpk,    // layer's packed W_post
                 const float* __restrict__ b_post,          // [64]
                 const float* __restrict__ amp, const float* __restrict__ att,
                 const float* __restrict__ snorm,           // [N]
                 float* __restrict__ hpost, float* __restrict__ bn) {
  int wave = threadIdx.x >> 5;
  int lane = threadIdx.x & 31;
  int tile = blockIdx.x * 8 + wave;
  int tbase = tile * 16;
  int half = lane >> 4;
  int m = lane & 15;

  const unsigned short* rh = hb + (size_t)(tbase + m) * 64;
  const unsigned short* rb = base + (size_t)(tbase + m) * 256;

  v8f acc0[4] = {}, accC[4] = {}, accD[4] = {};

  // h chunks (K = 0..63)
#pragma unroll
  for (int c = 0; c < 2; ++c) {
    const unsigned short* p = rh + c * 32;
    v8u16 lo = *(const v8u16*)(p + half * 8);
    v8u16 hi = *(const v8u16*)(p + 16 + half * 8);
    v16u16 a;
#pragma unroll
    for (int j = 0; j < 8; ++j) { a[j] = lo[j]; a[8 + j] = hi[j]; }
    v16bf av = __builtin_bit_cast(v16bf, a);
#pragma unroll
    for (int t = 0; t < 4; ++t) {
      v16u16 b = *(const v16u16*)(wpk + ((size_t)(t * 26 + c) * 32 + lane) * 16);
      acc0[t] = __builtin_amdgcn_wmma_f32_16x16x32_bf16(
          false, av, false, __builtin_bit_cast(v16bf, b), (short)0, acc0[t], false, false);
    }
  }
  // base chunks (reused for Wb / Wc / Wd)
#pragma unroll
  for (int cc = 0; cc < 8; ++cc) {
    const unsigned short* p = rb + cc * 32;
    v8u16 lo = *(const v8u16*)(p + half * 8);
    v8u16 hi = *(const v8u16*)(p + 16 + half * 8);
    v16u16 a;
#pragma unroll
    for (int j = 0; j < 8; ++j) { a[j] = lo[j]; a[8 + j] = hi[j]; }
    v16bf av = __builtin_bit_cast(v16bf, a);
#pragma unroll
    for (int t = 0; t < 4; ++t) {
      v16u16 b0 = *(const v16u16*)(wpk + ((size_t)(t * 26 + 2 + cc) * 32 + lane) * 16);
      v16u16 bc = *(const v16u16*)(wpk + ((size_t)(t * 26 + 10 + cc) * 32 + lane) * 16);
      v16u16 bd = *(const v16u16*)(wpk + ((size_t)(t * 26 + 18 + cc) * 32 + lane) * 16);
      acc0[t] = __builtin_amdgcn_wmma_f32_16x16x32_bf16(
          false, av, false, __builtin_bit_cast(v16bf, b0), (short)0, acc0[t], false, false);
      accC[t] = __builtin_amdgcn_wmma_f32_16x16x32_bf16(
          false, av, false, __builtin_bit_cast(v16bf, bc), (short)0, accC[t], false, false);
      accD[t] = __builtin_amdgcn_wmma_f32_16x16x32_bf16(
          false, av, false, __builtin_bit_cast(v16bf, bd), (short)0, accD[t], false, false);
    }
  }

  float av_[8], at_[8], sn_[8];
#pragma unroll
  for (int r = 0; r < 8; ++r) {
    int n2 = tbase + r + 8 * half;
    av_[r] = amp[n2];
    at_[r] = att[n2];
    sn_[r] = snorm[n2];
  }
#pragma unroll
  for (int t = 0; t < 4; ++t) {
    int feat = t * 16 + m;
    float bp = b_post[feat];
    float ls = 0.f, lss = 0.f;
#pragma unroll
    for (int r = 0; r < 8; ++r) {
      int n2 = tbase + r + 8 * half;
      float v = (acc0[t][r] + av_[r] * accC[t][r] + at_[r] * accD[t][r] + bp) * sn_[r];
      hpost[(size_t)n2 * 64 + feat] = v;
      ls += v;
      lss += v * v;
    }
    atom_add(&bn[feat], ls);
    atom_add(&bn[64 + feat], lss);
  }
}

// ---------------------------------------------------------------------------
// BatchNorm (train-mode, population var) + ReLU + residual. Updates h & h_bf16.
// ---------------------------------------------------------------------------
__global__ void __launch_bounds__(256)
bn_apply_kernel(const float* __restrict__ hpost, const float* __restrict__ bn,
                const float* __restrict__ gamma, const float* __restrict__ beta,
                float* __restrict__ h, unsigned short* __restrict__ hb) {
  int idx = blockIdx.x * 256 + threadIdx.x;          // N*64
  int f = idx & 63;
  float inv_n = 1.0f / (float)N_NODES;
  float mu = bn[f] * inv_n;
  float var = bn[64 + f] * inv_n - mu * mu;
  float x = hpost[idx];
  float y = gamma[f] * (x - mu) * rsqrtf(var + EPS_F) + beta[f];
  y = fmaxf(y, 0.0f) + h[idx];
  h[idx] = y;
  hb[idx] = f2u16(y);
}

// ---------------------------------------------------------------------------
// Readout: per-graph mean (contiguous 128-node graphs) + 64->32->16->1 MLP.
// ---------------------------------------------------------------------------
__global__ void __launch_bounds__(64)
readout_kernel(const float* __restrict__ h, const float* __restrict__ W1,
               const float* __restrict__ b1, const float* __restrict__ W2,
               const float* __restrict__ b2, const float* __restrict__ W3,
               const float* __restrict__ b3, float* __restrict__ out) {
  __shared__ float hg[64], o1[32], o2[16];
  int g = blockIdx.x;
  int f = threadIdx.x;
  float s = 0.f;
  const float* hp = h + (size_t)g * NODES_PER_GRAPH * 64 + f;
#pragma unroll 4
  for (int i = 0; i < NODES_PER_GRAPH; ++i) s += hp[i * 64];
  hg[f] = s * (1.0f / NODES_PER_GRAPH);
  __syncthreads();
  if (f < 32) {
    float a = b1[f];
    for (int k = 0; k < 64; ++k) a += hg[k] * W1[k * 32 + f];
    o1[f] = fmaxf(a, 0.f);
  }
  __syncthreads();
  if (f < 16) {
    float a = b2[f];
    for (int k = 0; k < 32; ++k) a += o1[k] * W2[k * 16 + f];
    o2[f] = fmaxf(a, 0.f);
  }
  __syncthreads();
  if (f == 0) {
    float a = b3[0];
    for (int k = 0; k < 16; ++k) a += o2[k] * W3[k];
    out[g] = a;
  }
}

// ---------------------------------------------------------------------------
extern "C" void kernel_launch(void* const* d_in, const int* in_sizes, int n_in,
                              void* d_out, int out_size, void* d_ws, size_t ws_size,
                              hipStream_t stream) {
  const int* h_tok = (const int*)d_in[0];
  const int* e_tok = (const int*)d_in[1];
  const int* src = (const int*)d_in[2];
  const int* dst = (const int*)d_in[3];
  /* graph_id (d_in[4]) unused: graphs are contiguous 128-node blocks */
  const float* snorm = (const float*)d_in[5];
  const float* emb_h = (const float*)d_in[6];
  const float* emb_e = (const float*)d_in[7];
  const float* W_pre = (const float*)d_in[8];
  const float* b_pre = (const float*)d_in[9];
  const float* W_post = (const float*)d_in[10];
  const float* b_post = (const float*)d_in[11];
  const float* bn_gamma = (const float*)d_in[12];
  const float* bn_beta = (const float*)d_in[13];
  const float* W1 = (const float*)d_in[14];
  const float* b1 = (const float*)d_in[15];
  const float* W2 = (const float*)d_in[16];
  const float* b2 = (const float*)d_in[17];
  const float* W3 = (const float*)d_in[18];
  const float* b3 = (const float*)d_in[19];
  float* out = (float*)d_out;

  // Workspace carve (256B aligned slots).
  char* p = (char*)d_ws;
  auto take = [&p](size_t bytes) {
    char* r = p;
    p += (bytes + 255) & ~(size_t)255;
    return r;
  };
  const size_t NF = (size_t)N_NODES * 64;
  float* h_f32 = (float*)take(NF * 4);
  unsigned short* h_bf = (unsigned short*)take(NF * 2);
  float* s1 = (float*)take(NF * 4);
  float* s2 = (float*)take(NF * 4);
  float* mxb = (float*)take(NF * 4);
  float* mnb = (float*)take(NF * 4);
  unsigned short* base = (unsigned short*)take((size_t)N_NODES * 256 * 2);
  float* hpost = (float*)take(NF * 4);
  float* deg = (float*)take((size_t)N_NODES * 4);
  float* amp = (float*)take((size_t)N_NODES * 4);
  float* att = (float*)take((size_t)N_NODES * 4);
  float* bn = (float*)take(128 * 4);
  unsigned short* wpre_pk = (unsigned short*)take((size_t)LAYERS * 4 * 6 * 512 * 2);
  unsigned short* wpost_pk = (unsigned short*)take((size_t)LAYERS * 4 * 26 * 512 * 2);

  // --- one-time setup ---
  pack_wpre_kernel<<<(LAYERS * 4 * 6 * 512 + 255) / 256, 256, 0, stream>>>(W_pre, wpre_pk);
  pack_wpost_kernel<<<(LAYERS * 4 * 26 * 512 + 255) / 256, 256, 0, stream>>>(W_post, wpost_pk);
  embed_kernel<<<NF / 256, 256, 0, stream>>>(h_tok, emb_h, h_f32, h_bf);
  deg_zero_kernel<<<N_NODES / 256, 256, 0, stream>>>(deg);
  deg_count_kernel<<<N_EDGES / 256, 256, 0, stream>>>(dst, deg);
  deg_scalars_kernel<<<N_NODES / 256, 256, 0, stream>>>(deg, amp, att);

  // --- PNA layers ---
  for (int l = 0; l < LAYERS; ++l) {
    const unsigned short* wpre_l = wpre_pk + (size_t)l * 4 * 6 * 512;
    const unsigned short* wpost_l = wpost_pk + (size_t)l * 4 * 26 * 512;
    layer_init_kernel<<<NF / 256, 256, 0, stream>>>(s1, s2, mxb, mnb, bn);
    edge_kernel<<<N_EDGES / (16 * 8), 256, 0, stream>>>(
        h_bf, src, dst, e_tok, emb_e, wpre_l, b_pre + l * 64, s1, s2, mxb, mnb);
    node_prep_kernel<<<NF / 256, 256, 0, stream>>>(s1, s2, mxb, mnb, deg, base);
    node_gemm_kernel<<<N_NODES / (16 * 8), 256, 0, stream>>>(
        h_bf, base, wpost_l, b_post + l * 64, amp, att, snorm, hpost, bn);
    bn_apply_kernel<<<NF / 256, 256, 0, stream>>>(
        hpost, bn, bn_gamma + l * 64, bn_beta + l * 64, h_f32, h_bf);
  }

  // --- readout ---
  readout_kernel<<<N_GRAPH, 64, 0, stream>>>(h_f32, W1, b1, W2, b2, W3, b3, out);
}
